// SGCrossAttention_2963527434559
// MI455X (gfx1250) — matrix operand order repaired and checked by reading, not compiled, not run on hardware
//
#include <hip/hip_runtime.h>

// ---------------------------------------------------------------------------
// SG cross-attention, fused for gfx1250 (MI455X): wave32, v_wmma_f32_16x16x32_f16,
// async global->LDS K/V staging (ASYNCcnt), double-buffered across heads.
// ---------------------------------------------------------------------------

typedef __attribute__((ext_vector_type(16))) _Float16 v16h;
typedef __attribute__((ext_vector_type(8)))  float    v8f;

union AFrag { unsigned int u[8]; v16h v; };

#define Z8 {0.f, 0.f, 0.f, 0.f, 0.f, 0.f, 0.f, 0.f}

// LDS layout (dynamic block, no static LDS => block starts at LDS offset 0):
//   [0,128K)    xhat 64x1024 f16   (aliased as out 64x512 f16 after phase B)
//   [128K,192K) q    64x512  f16
//   [192K,224K) k/v buffer 0 : k_h 64x128 f16 (dh-major) + v_h 128x64 f16
//   [224K,256K) k/v buffer 1
//   [256K,288K) scores 64x128 f32
//   [288K,304K) attn 64x128 f16
#define LDS_XHAT   0u
#define LDS_Q      131072u
#define LDS_KV     196608u
#define LDS_KVSTR  32768u     // per double-buffer stage (16K k + 16K v)
#define LDS_SCO    262144u
#define LDS_ATTN   294912u
#define LDS_TOTAL  311296u    // 304 KB of the 320 KB WGP LDS

// --- WMMA wrapper: D = A(16x32 f16) * B(32x16 f16) + C(16x16 f32) ---
__device__ __forceinline__ v8f wmma16(v16h a, v16h b, v8f c) {
  return __builtin_amdgcn_wmma_f32_16x16x32_f16(
      /*neg_a=*/false, a, /*neg_b=*/false, b,
      /*c_mod=*/(short)0, c, /*reuse_a=*/false, /*reuse_b=*/false);
}

// A-fragment (16x32, row-major f16 source, stride lda in halves).
__device__ __forceinline__ v16h load_a(const _Float16* base, int lda, int lane) {
  int row = lane & 15;
  int kb  = (lane >> 4) << 4;        // 0 or 16
  const _Float16* r = base + row * lda;
  AFrag f;
#pragma unroll
  for (int j = 0; j < 8; ++j) {
    int ko = kb + (j << 1);
    f.u[j] = *(const unsigned int*)(r + ko);
  }
  return f.v;
}

// B-fragment (32x16, K-major f16 source, stride ldb in halves).
__device__ __forceinline__ v16h load_b(const _Float16* base, int ldb, int lane) {
  int n  = lane & 15;
  int kb = (lane >> 4) << 4;         // 0 or 16
  AFrag f;
#pragma unroll
  for (int j = 0; j < 8; ++j) {
    union { _Float16 h[2]; unsigned int u; } p;
    p.h[0] = base[(kb + 2 * j)     * ldb + n];
    p.h[1] = base[(kb + 2 * j + 1) * ldb + n];
    f.u[j] = p.u;
  }
  return f.v;
}

// B-fragment from pre-swizzled weights: contiguous 32 B per lane (2x b128).
__device__ __forceinline__ v16h load_b_sw(const _Float16* tile, int lane) {
  const uint4* p = (const uint4*)(tile + lane * 16);
  uint4 q0 = p[0], q1 = p[1];
  AFrag f;
  f.u[0] = q0.x; f.u[1] = q0.y; f.u[2] = q0.z; f.u[3] = q0.w;
  f.u[4] = q1.x; f.u[5] = q1.y; f.u[6] = q1.z; f.u[7] = q1.w;
  return f.v;
}

// Async global->LDS 16B copy, tracked by ASYNCcnt (CDNA5).
__device__ __forceinline__ void async_b128(unsigned lds_byte_off, const void* g) {
  asm volatile("global_load_async_to_lds_b128 %0, %1, off"
               :: "v"(lds_byte_off), "v"((unsigned long long)(size_t)g)
               : "memory");
}

// Issue one head's k/v (2 x 16 KB) into double-buffer `buf`: 8 ops per thread.
__device__ __forceinline__ void stage_kv(const _Float16* kT, const _Float16* vT,
                                         int b, int h, int buf, int tid) {
  const char* kg = (const char*)(kT + ((size_t)(b * 8 + h)) * 8192);
  const char* vg = (const char*)(vT + ((size_t)(b * 8 + h)) * 8192);
  unsigned kOff = LDS_KV + (unsigned)buf * LDS_KVSTR;
  unsigned vOff = kOff + 16384u;
#pragma unroll
  for (int i = 0; i < 4; ++i) {
    unsigned off = ((unsigned)tid + 256u * i) * 16u;
    async_b128(kOff + off, kg + off);
    async_b128(vOff + off, vg + off);
  }
}

// ---------------------------------------------------------------------------
// Prep kernel 1: weights fp32 -> f16; Wq/Wo pre-swizzled to B-fragment order.
// ---------------------------------------------------------------------------
__global__ void prep_weights(const float* __restrict__ Wq, const float* __restrict__ Wk,
                             const float* __restrict__ Wv, const float* __restrict__ Wo,
                             _Float16* __restrict__ Wq_sw, _Float16* __restrict__ Wk16,
                             _Float16* __restrict__ Wv16, _Float16* __restrict__ Wo_sw) {
  int id = blockIdx.x * 256 + threadIdx.x;
  if (id < 524288) {                 // Wq_sw: K=1024 (32 ksteps), N=512 (32 ntiles)
    int o = id, tile = o >> 9, w = o & 511;
    int lane = w >> 4, idx = w & 15;
    int j = idx >> 1, half = idx & 1;
    int n_local = lane & 15, kb = (lane >> 4) << 4;
    int k_local = kb + 2 * j + half;
    int ks = tile >> 5, nt = tile & 31;
    Wq_sw[o] = (_Float16)Wq[(ks * 32 + k_local) * 512 + nt * 16 + n_local];
  } else if (id < 1048576) {         // Wo_sw: K=512 (16 ksteps), N=1024 (64 ntiles)
    int o = id - 524288, tile = o >> 9, w = o & 511;
    int lane = w >> 4, idx = w & 15;
    int j = idx >> 1, half = idx & 1;
    int n_local = lane & 15, kb = (lane >> 4) << 4;
    int k_local = kb + 2 * j + half;
    int ks = tile >> 6, nt = tile & 63;
    Wo_sw[o] = (_Float16)Wo[(ks * 32 + k_local) * 1024 + nt * 16 + n_local];
  } else if (id < 1441792) {
    int o = id - 1048576; Wk16[o] = (_Float16)Wk[o];
  } else if (id < 1835008) {
    int o = id - 1441792; Wv16[o] = (_Float16)Wv[o];
  }
}

// ---------------------------------------------------------------------------
// Prep kernel 2: LayerNorm over context rows (512 rows x 768) -> f16
// ---------------------------------------------------------------------------
__global__ void ln_ctx(const float* __restrict__ ctx, const float* __restrict__ gk,
                       const float* __restrict__ bk, _Float16* __restrict__ ck) {
  int row = blockIdx.x, tid = threadIdx.x;
  const float* r = ctx + (size_t)row * 768;
  float s = 0.f, s2 = 0.f;
  for (int i = tid; i < 768; i += 256) { float t = r[i]; s += t; s2 += t * t; }
  for (int m = 16; m; m >>= 1) { s += __shfl_xor(s, m); s2 += __shfl_xor(s2, m); }
  __shared__ float ss[8], sq[8];
  int wave = tid >> 5, lane = tid & 31;
  if (lane == 0) { ss[wave] = s; sq[wave] = s2; }
  __syncthreads();
  if (tid == 0) {
    float a = 0.f, b = 0.f;
    for (int i = 0; i < 8; ++i) { a += ss[i]; b += sq[i]; }
    ss[0] = a; sq[0] = b;
  }
  __syncthreads();
  float mean = ss[0] * (1.f / 768.f);
  float var  = sq[0] * (1.f / 768.f) - mean * mean;
  float rs   = rsqrtf(var + 1e-5f);
  for (int i = tid; i < 768; i += 256) {
    float t = r[i];
    ck[(size_t)row * 768 + i] = (_Float16)(((t - mean) * rs) * gk[i] + bk[i]);
  }
}

// ---------------------------------------------------------------------------
// Prep kernel 3: k,v projections, stored head-major:
//   kT[(b*8+h)*64+dh][m]  (dh-major, B-ready for q.k^T)
//   vT[(b*8+h)*128+m][dh] (m-major,  B-ready for attn.v)
// ---------------------------------------------------------------------------
__global__ void kv_proj(const _Float16* __restrict__ ck, const _Float16* __restrict__ Wk16,
                        const _Float16* __restrict__ Wv16,
                        _Float16* __restrict__ kT, _Float16* __restrict__ vT) {
  int id = blockIdx.x * 256 + threadIdx.x;            // 0 .. 524287
  bool isV = id >= 262144;
  int o = isV ? id - 262144 : id;
  int m, dh, h, b;
  if (!isV) { m = o & 127; int r = o >> 7; dh = r & 63; r >>= 6; h = r & 7; b = r >> 3; }
  else      { dh = o & 63; int r = o >> 6; m = r & 127; r >>= 7; h = r & 7; b = r >> 3; }
  int col = h * 64 + dh;
  const _Float16* crow = ck + ((size_t)(b * 128 + m)) * 768;
  const _Float16* W = isV ? Wv16 : Wk16;
  float acc = 0.f;
  for (int c = 0; c < 768; ++c) acc += (float)crow[c] * (float)W[c * 512 + col];
  if (!isV) kT[o] = (_Float16)acc; else vT[o] = (_Float16)acc;
}

// ---------------------------------------------------------------------------
// Prep kernel 4: per-(bt,m) projected coords u,v and raw camera z.
// ---------------------------------------------------------------------------
__global__ void bias_params(const float* __restrict__ pos, const float* __restrict__ c2w,
                            const float* __restrict__ intr,
                            float* __restrict__ U, float* __restrict__ V, float* __restrict__ Z) {
  int id = blockIdx.x * 256 + threadIdx.x;
  if (id >= 4096) return;
  int m = id & 127, bt = id >> 7, b = bt >> 3, t = bt & 7;
  const float* M4 = c2w + (size_t)(b * 8 + t) * 16;
  float px = pos[(b * 128 + m) * 3 + 0];
  float py = pos[(b * 128 + m) * 3 + 1];
  float pz = pos[(b * 128 + m) * 3 + 2];
  float dx = px - M4[3], dy = py - M4[7], dz = pz - M4[11];
  float cx_ = M4[0] * dx + M4[4] * dy + M4[8]  * dz;   // R^T (p - t)
  float cy_ = M4[1] * dx + M4[5] * dy + M4[9]  * dz;
  float cz_ = M4[2] * dx + M4[6] * dy + M4[10] * dz;
  float z  = fmaxf(cz_, 1e-4f);
  float fx = intr[b * 9 + 0], fy = intr[b * 9 + 4];
  float cxi = intr[b * 9 + 2], cyi = intr[b * 9 + 5];
  U[id] = fx * (cx_ / z) + cxi;
  V[id] = fy * (cy_ / z) + cyi;
  Z[id] = cz_;
}

// ---------------------------------------------------------------------------
// Main fused kernel: one workgroup per (bt, 64-row patch tile). 256 thr = 8 waves.
// ---------------------------------------------------------------------------
__global__ void __launch_bounds__(256)
sg_attn_main(const float* __restrict__ x, const float* __restrict__ gq,
             const float* __restrict__ bq,
             const _Float16* __restrict__ Wq_sw, const _Float16* __restrict__ Wo_sw,
             const _Float16* __restrict__ kT, const _Float16* __restrict__ vT,
             const float* __restrict__ U, const float* __restrict__ Vb,
             const float* __restrict__ Zb,
             const float* __restrict__ slog, const float* __restrict__ dalpha,
             const float* __restrict__ gate, const float* __restrict__ sgate,
             float* __restrict__ out) {
  extern __shared__ char smem[];
  _Float16* xhat = (_Float16*)(smem + LDS_XHAT);
  _Float16* outb = (_Float16*)(smem + LDS_XHAT);       // alias after phase B
  _Float16* qbuf = (_Float16*)(smem + LDS_Q);
  float*    sco  = (float*)   (smem + LDS_SCO);
  _Float16* attn = (_Float16*)(smem + LDS_ATTN);

  const int tid = threadIdx.x, wave = tid >> 5, lane = tid & 31;
  const int bt = blockIdx.x / 36, tileIdx = blockIdx.x % 36;
  const int p0 = tileIdx * 64;
  const int b  = bt >> 3;
  const float tg  = tanhf(gate[0]);
  const float tgs = tanhf(sgate[0]);

  // Kick off head 0's k/v async staging: overlaps the whole LN + Wq GEMM.
  stage_kv(kT, vT, b, 0, 0, tid);

  // ---- Phase A: LayerNorm(x) rows -> xhat f16 (8 rows per wave) ----
  for (int r = 0; r < 8; ++r) {
    int row = wave * 8 + r;
    const float* xr = x + ((size_t)(bt * 2304 + p0 + row)) * 1024;
    __builtin_prefetch(xr + lane * 32, 0, 3);
    float s = 0.f, s2 = 0.f;
#pragma unroll 4
    for (int i = 0; i < 32; ++i) { float t = xr[lane + 32 * i]; s += t; s2 += t * t; }
    for (int m = 16; m; m >>= 1) { s += __shfl_xor(s, m); s2 += __shfl_xor(s2, m); }
    float mean = s * (1.f / 1024.f);
    float var  = s2 * (1.f / 1024.f) - mean * mean;
    float rs   = rsqrtf(var + 1e-5f);
#pragma unroll 4
    for (int i = 0; i < 32; ++i) {
      int c = lane + 32 * i;
      float t = xr[c];
      xhat[row * 1024 + c] = (_Float16)(((t - mean) * rs) * gq[c] + bq[c]);
    }
  }
  __syncthreads();

  // ---- Phase B: q = xhat @ Wq (64x1024 . 1024x512) ----
  // Per wave: fixed mt, 16 n-tiles processed in pairs sharing one A fragment.
  {
    int mt  = wave >> 1;
    int ntB = (wave & 1) * 16;
    for (int t2 = 0; t2 < 8; ++t2) {
      int nt0 = ntB + 2 * t2, nt1 = nt0 + 1;
      v8f acc0 = Z8, acc1 = Z8;
      for (int ks = 0; ks < 32; ++ks) {
        v16h a  = load_a(xhat + mt * 16 * 1024 + ks * 32, 1024, lane);
        v16h b0 = load_b_sw(Wq_sw + (((size_t)ks * 32 + nt0) << 9), lane);
        v16h b1 = load_b_sw(Wq_sw + (((size_t)ks * 32 + nt1) << 9), lane);
        acc0 = wmma16(a, b0, acc0);
        acc1 = wmma16(a, b1, acc1);
      }
      int r0 = mt * 16 + ((lane >> 4) << 3);
      int c0 = nt0 * 16 + (lane & 15), c1 = nt1 * 16 + (lane & 15);
#pragma unroll
      for (int j = 0; j < 8; ++j) {
        qbuf[(r0 + j) * 512 + c0] = (_Float16)acc0[j];
        qbuf[(r0 + j) * 512 + c1] = (_Float16)acc1[j];
      }
    }
  }
  __syncthreads();

  // ---- Per-head attention, k/v double-buffered via ASYNCcnt ----
  for (int h = 0; h < 8; ++h) {
    float sig     = __expf(slog[h]);
    float inv2s_h = 1.f / (2.f * sig * sig + 1e-8f);
    float alpha   = dalpha[h];

    _Float16* kbuf = (_Float16*)(smem + LDS_KV + (unsigned)(h & 1) * LDS_KVSTR);
    _Float16* vbuf = kbuf + 8192;

    if (h < 7) {
      stage_kv(kT, vT, b, h + 1, (h + 1) & 1, tid);     // prefetch next head
      asm volatile("s_wait_asynccnt 0x8" ::: "memory"); // this head's 8 are done
    } else {
      asm volatile("s_wait_asynccnt 0x0" ::: "memory");
    }
    __syncthreads();

    // scores = q_h . k_h^T (64x64 . 64x128) + fused spatial bias
    for (int t = 0; t < 4; ++t) {
      int mt = wave >> 1, nt = (wave & 1) * 4 + t;
      v8f acc = Z8;
      for (int ks = 0; ks < 2; ++ks) {
        v16h a   = load_a(qbuf + mt * 16 * 512 + h * 64 + ks * 32, 512, lane);
        v16h bfr = load_b(kbuf + ks * 32 * 128 + nt * 16, 128, lane);
        acc = wmma16(a, bfr, acc);
      }
      int mcol = nt * 16 + (lane & 15);
      int r0   = mt * 16 + ((lane >> 4) << 3);
      float u  = U[bt * 128 + mcol];
      float vv = Vb[bt * 128 + mcol];
      float zr = Zb[bt * 128 + mcol];
      float z  = fmaxf(zr, 1e-4f);
      float i2 = z * z * inv2s_h;
      float dterm  = alpha * (-__logf(z));
      bool  behind = (zr <= 0.f);
#pragma unroll
      for (int j = 0; j < 8; ++j) {
        int pl = r0 + j;
        int pg = p0 + pl;
        float gx = ((float)(pg % 48) + 0.5f) * (1.f / 48.f);
        float gy = ((float)(pg / 48) + 0.5f) * (1.f / 48.f);
        float ddx = gx - u, ddy = gy - vv;
        float bias = behind ? 0.f : (-(ddx * ddx + ddy * ddy) * i2 + dterm);
        sco[pl * 128 + mcol] = acc[j] * 0.125f + tgs * bias;
      }
    }
    __syncthreads();

    // softmax over M=128: 4 lanes per row, fully in-wave
    {
      int row = tid >> 2, sub = tid & 3;
      float mx = -3.0e38f;
#pragma unroll 4
      for (int i = 0; i < 32; ++i) mx = fmaxf(mx, sco[row * 128 + sub + 4 * i]);
      mx = fmaxf(mx, __shfl_xor(mx, 1));
      mx = fmaxf(mx, __shfl_xor(mx, 2));
      float sum = 0.f;
#pragma unroll 4
      for (int i = 0; i < 32; ++i) {
        int c = sub + 4 * i;
        float e = __expf(sco[row * 128 + c] - mx);
        sum += e;
        sco[row * 128 + c] = e;
      }
      sum += __shfl_xor(sum, 1);
      sum += __shfl_xor(sum, 2);
      float rinv = 1.f / sum;
#pragma unroll 4
      for (int i = 0; i < 32; ++i) {
        int c = sub + 4 * i;
        attn[row * 128 + c] = (_Float16)(sco[row * 128 + c] * rinv);
      }
    }
    __syncthreads();

    // out_h = attn . v_h (64x128 . 128x64): one n-tile pair per wave
    {
      int mt = wave >> 1, nt0 = (wave & 1) * 2, nt1 = nt0 + 1;
      v8f acc0 = Z8, acc1 = Z8;
      for (int ks = 0; ks < 4; ++ks) {
        v16h a  = load_a(attn + mt * 16 * 128 + ks * 32, 128, lane);
        v16h b0 = load_b(vbuf + ks * 32 * 64 + nt0 * 16, 64, lane);
        v16h b1 = load_b(vbuf + ks * 32 * 64 + nt1 * 16, 64, lane);
        acc0 = wmma16(a, b0, acc0);
        acc1 = wmma16(a, b1, acc1);
      }
      int r0 = mt * 16 + ((lane >> 4) << 3);
      int c0 = h * 64 + nt0 * 16 + (lane & 15);
      int c1 = h * 64 + nt1 * 16 + (lane & 15);
#pragma unroll
      for (int j = 0; j < 8; ++j) {
        outb[(r0 + j) * 512 + c0] = (_Float16)acc0[j];
        outb[(r0 + j) * 512 + c1] = (_Float16)acc1[j];
      }
    }
    __syncthreads();
  }

  // ---- Phase E: y = out @ Wo (64x512 . 512x1024), residual + gate ----
  {
    int mt  = wave >> 1;
    int ntB = (wave & 1) * 32;
    for (int t2 = 0; t2 < 16; ++t2) {
      int nt0 = ntB + 2 * t2, nt1 = nt0 + 1;
      v8f acc0 = Z8, acc1 = Z8;
      for (int ks = 0; ks < 16; ++ks) {
        v16h a  = load_a(outb + mt * 16 * 512 + ks * 32, 512, lane);
        v16h b0 = load_b_sw(Wo_sw + (((size_t)ks * 64 + nt0) << 9), lane);
        v16h b1 = load_b_sw(Wo_sw + (((size_t)ks * 64 + nt1) << 9), lane);
        acc0 = wmma16(a, b0, acc0);
        acc1 = wmma16(a, b1, acc1);
      }
      int r0 = mt * 16 + ((lane >> 4) << 3);
      int c0 = nt0 * 16 + (lane & 15), c1 = nt1 * 16 + (lane & 15);
#pragma unroll
      for (int j = 0; j < 8; ++j) {
        size_t gi0 = ((size_t)(bt * 2304 + p0 + r0 + j)) * 1024 + c0;
        size_t gi1 = ((size_t)(bt * 2304 + p0 + r0 + j)) * 1024 + c1;
        out[gi0] = x[gi0] + tg * acc0[j];
        out[gi1] = x[gi1] + tg * acc1[j];
      }
    }
  }
}

// ---------------------------------------------------------------------------
// Host launcher
// ---------------------------------------------------------------------------
extern "C" void kernel_launch(void* const* d_in, const int* in_sizes, int n_in,
                              void* d_out, int out_size, void* d_ws, size_t ws_size,
                              hipStream_t stream) {
  (void)in_sizes; (void)n_in; (void)out_size; (void)ws_size;
  const float* x    = (const float*)d_in[0];
  const float* ctx  = (const float*)d_in[1];
  const float* pos  = (const float*)d_in[2];
  const float* c2w  = (const float*)d_in[3];
  const float* intr = (const float*)d_in[4];
  const float* Wq   = (const float*)d_in[5];
  const float* Wk   = (const float*)d_in[6];
  const float* Wv   = (const float*)d_in[7];
  const float* Wo   = (const float*)d_in[8];
  const float* gate = (const float*)d_in[9];
  const float* sg   = (const float*)d_in[10];
  const float* slog = (const float*)d_in[11];
  const float* dal  = (const float*)d_in[12];
  const float* gq   = (const float*)d_in[13];
  const float* bq   = (const float*)d_in[14];
  const float* gk   = (const float*)d_in[15];
  const float* bk   = (const float*)d_in[16];

  char* ws = (char*)d_ws;
  _Float16* Wq_sw = (_Float16*)(ws + 0);          // 1,048,576 B
  _Float16* Wk16  = (_Float16*)(ws + 1048576);    //   786,432 B
  _Float16* Wv16  = (_Float16*)(ws + 1835008);    //   786,432 B
  _Float16* Wo_sw = (_Float16*)(ws + 2621440);    // 1,048,576 B
  _Float16* ck16  = (_Float16*)(ws + 3670016);    //   786,432 B
  _Float16* kT    = (_Float16*)(ws + 4456448);    //   524,288 B
  _Float16* vT    = (_Float16*)(ws + 4980736);    //   524,288 B
  float*    U     = (float*)   (ws + 5505024);    //    16,384 B
  float*    V     = (float*)   (ws + 5521408);    //    16,384 B
  float*    Z     = (float*)   (ws + 5537792);    //    16,384 B

  prep_weights<<<7168, 256, 0, stream>>>(Wq, Wk, Wv, Wo, Wq_sw, Wk16, Wv16, Wo_sw);
  ln_ctx<<<512, 256, 0, stream>>>(ctx, gk, bk, ck16);
  kv_proj<<<2048, 256, 0, stream>>>(ck16, Wk16, Wv16, kT, vT);
  bias_params<<<16, 256, 0, stream>>>(pos, c2w, intr, U, V, Z);

  (void)hipFuncSetAttribute(reinterpret_cast<const void*>(&sg_attn_main),
                            hipFuncAttributeMaxDynamicSharedMemorySize,
                            (int)LDS_TOTAL);
  sg_attn_main<<<1152, 256, LDS_TOTAL, stream>>>(x, gq, bq, Wq_sw, Wo_sw, kT, vT,
                                                 U, V, Z, slog, dal, gate, sg,
                                                 (float*)d_out);
}
